// MineModel_16776142258483
// MI455X (gfx1250) — matrix operand order repaired and checked by reading, not compiled, not run on hardware
//
#include <hip/hip_runtime.h>

#define QLEN  10000
#define TLEN  3000
#define EMB   768
#define HEADS 4
#define DK    192
#define HNEED 3
#define NQ    (QLEN*HNEED)   /* 30000 */
#define EG    100000
#define EI    100000
#define EG3   (EG*HNEED)     /* 300000 */
#define NCLUST 10

typedef __attribute__((ext_vector_type(16))) __bf16 v16bf;
typedef __attribute__((ext_vector_type(8)))  float  v8f;
typedef __attribute__((ext_vector_type(4)))  unsigned int v4u;
typedef __attribute__((ext_vector_type(8)))  int v8i;
typedef __attribute__((ext_vector_type(4)))  int v4i;
typedef unsigned int   u32;
typedef unsigned short u16;

// ---------------- small device helpers ----------------
__device__ __forceinline__ u16 f2bf(float x) {
    u32 u = __float_as_uint(x);
    u32 r = u + 0x7FFFu + ((u >> 16) & 1u);   // round-to-nearest-even
    return (u16)(r >> 16);
}
// order-preserving float -> uint key (for atomicMax-based segment max)
__device__ __forceinline__ u32 fkey(float f) {
    u32 u = __float_as_uint(f);
    return (u & 0x80000000u) ? ~u : (u | 0x80000000u);
}
__device__ __forceinline__ float kdec(u32 k) {
    u32 u = (k & 0x80000000u) ? (k & 0x7FFFFFFFu) : ~k;
    return __uint_as_float(u);
}

// ---------------- elementwise kernels ----------------
__global__ void k_f32_to_bf16(const float* __restrict__ in, u16* __restrict__ o, long n) {
    long i = (long)blockIdx.x * blockDim.x + threadIdx.x;
    if (i < n) o[i] = f2bf(in[i]);
}
__global__ void k_fill_u32(u32* __restrict__ p, u32 v, long n) {
    long i = (long)blockIdx.x * blockDim.x + threadIdx.x;
    if (i < n) p[i] = v;
}
__global__ void k_copy_f32(const float* __restrict__ a, float* __restrict__ b, long n) {
    long i = (long)blockIdx.x * blockDim.x + threadIdx.x;
    if (i < n) b[i] = a[i];
}
__global__ void k_add_inplace(float* __restrict__ y, const float* __restrict__ x, long n) {
    long i = (long)blockIdx.x * blockDim.x + threadIdx.x;
    if (i < n) y[i] += x[i];
}
__global__ void k_avg_add(float* __restrict__ h, const float* __restrict__ a,
                          const float* __restrict__ b, long n) {
    long i = (long)blockIdx.x * blockDim.x + threadIdx.x;
    if (i < n) h[i] += 0.5f * (a[i] + b[i]);
}

// q_h[(hd*Q+qi), f] = q_hs[qi, hd*EMB+f] + q_emb[qi, f]
__global__ void k_build_qh(const float* __restrict__ qhs, const float* __restrict__ qemb,
                           float* __restrict__ qh) {
    long i = (long)blockIdx.x * blockDim.x + threadIdx.x;
    if (i >= (long)NQ * EMB) return;
    long row = i / EMB; int f = (int)(i % EMB);
    int hd = (int)(row / QLEN), qi = (int)(row % QLEN);
    qh[i] = qhs[(long)qi * (EMB * HNEED) + (long)hd * EMB + f] + qemb[(long)qi * EMB + f];
}

// 10-way cluster assignment softmax; Wc staged in LDS
__global__ __launch_bounds__(256)
void k_assign(const float* __restrict__ qh, const float* __restrict__ Wc,
              const float* __restrict__ bc, float* __restrict__ assign) {
    __shared__ float sW[EMB * NCLUST];  // 30 KB
    for (int i = threadIdx.x; i < EMB * NCLUST; i += 256) sW[i] = Wc[i];
    __syncthreads();
    long row = (long)blockIdx.x * 256 + threadIdx.x;
    if (row >= NQ) return;
    float a[NCLUST];
#pragma unroll
    for (int c = 0; c < NCLUST; c++) a[c] = bc[c];
    const float* x = qh + row * EMB;
    for (int k = 0; k < EMB; k++) {
        float xv = x[k];
        const float* wr = &sW[k * NCLUST];
#pragma unroll
        for (int c = 0; c < NCLUST; c++) a[c] += xv * wr[c];
    }
    float m = a[0];
#pragma unroll
    for (int c = 1; c < NCLUST; c++) m = fmaxf(m, a[c]);
    float s = 0.f;
#pragma unroll
    for (int c = 0; c < NCLUST; c++) { a[c] = __expf(a[c] - m); s += a[c]; }
    float inv = 1.f / s;
#pragma unroll
    for (int c = 0; c < NCLUST; c++) assign[row * NCLUST + c] = a[c] * inv;
}

__global__ void k_centers_add(const float* __restrict__ assign, const float* __restrict__ centers,
                              float* __restrict__ qh) {
    long i = (long)blockIdx.x * blockDim.x + threadIdx.x;
    if (i >= (long)NQ * EMB) return;
    long row = i / EMB; int f = (int)(i % EMB);
    float s = 0.f;
#pragma unroll
    for (int c = 0; c < NCLUST; c++) s += assign[row * NCLUST + c] * centers[c * EMB + f];
    qh[i] += s;
}

// replicate give edges per model-level head
__global__ void k_make_edges(const int* __restrict__ gsrc, const int* __restrict__ gdst,
                             int* __restrict__ osrc, int* __restrict__ odst) {
    long i = (long)blockIdx.x * blockDim.x + threadIdx.x;
    if (i >= EG3) return;
    int base = (int)(i % EG), rep = (int)(i / EG);
    osrc[i] = gsrc[base];
    odst[i] = gdst[base] + rep * QLEN;
}

// ---------------- edge attention kernels ----------------
__global__ void k_edge_score(const int* __restrict__ src, const int* __restrict__ dst, long E,
                             const float* __restrict__ qmat, const float* __restrict__ kmat,
                             const float* __restrict__ rpri, float* __restrict__ score,
                             u32* __restrict__ mkey) {
    long t = (long)blockIdx.x * blockDim.x + threadIdx.x;
    if (t >= E * HEADS) return;
    long e = t >> 2; int h = (int)(t & 3);
    int s = src[e], d = dst[e];
    const float4* qp = (const float4*)(qmat + (long)d * EMB + h * DK);
    const float4* kp = (const float4*)(kmat + (long)s * EMB + h * DK);
    float acc = 0.f;
#pragma unroll 4
    for (int i = 0; i < DK / 4; i++) {
        float4 a = qp[i], b = kp[i];
        acc += a.x * b.x + a.y * b.y + a.z * b.z + a.w * b.w;
    }
    acc *= rpri[h] * 0.0721687836487f;  // 1/sqrt(192)
    score[t] = acc;
    atomicMax(&mkey[(long)d * HEADS + h], fkey(acc));
}

__global__ void k_edge_exp(const int* __restrict__ dst, long E,
                           const float* __restrict__ score, const u32* __restrict__ mkey,
                           float* __restrict__ att, float* __restrict__ ssum) {
    long t = (long)blockIdx.x * blockDim.x + threadIdx.x;
    if (t >= E * HEADS) return;
    long e = t >> 2; int h = (int)(t & 3);
    int d = dst[e];
    float m = kdec(mkey[(long)d * HEADS + h]);
    float a = __expf(score[t] - m);
    att[t] = a;
    atomicAdd(&ssum[(long)d * HEADS + h], a);
}

__global__ void k_edge_scatter(const int* __restrict__ src, const int* __restrict__ dst, long E,
                               const float* __restrict__ att, const float* __restrict__ ssum,
                               const float* __restrict__ vmat, float* __restrict__ outm) {
    long t = (long)blockIdx.x * blockDim.x + threadIdx.x;
    if (t >= E * EMB) return;
    long e = t / EMB; int j = (int)(t % EMB); int h = j / DK;
    int s = src[e], d = dst[e];
    float w = att[e * HEADS + h] / ssum[(long)d * HEADS + h];
    atomicAdd(&outm[(long)d * EMB + j], vmat[(long)s * EMB + j] * w);
}

// ques[qi][hd][f] = ques_heads[hd*Q+qi][f]
__global__ void k_write_ques(const float* __restrict__ qheads, float* __restrict__ outp) {
    long i = (long)blockIdx.x * blockDim.x + threadIdx.x;
    if (i >= (long)QLEN * HNEED * EMB) return;
    long qi = i / (HNEED * EMB);
    long rem = i % (HNEED * EMB);
    int hd = (int)(rem / EMB), f = (int)(rem % EMB);
    outp[i] = qheads[((long)hd * QLEN + qi) * EMB + f];
}

// ---------------- WMMA bf16 GEMM: C = A(bf16) @ B(bf16) + bias ----------------
// Block tile 64x128, K-step 32, 256 threads = 8 wave32s in a 4x2 grid.
// Each wave: one 16-row strip, four 16x16 output tiles -> 4 WMMAs sharing one
// A fragment per K-step; K compile-time => unroll-by-2 (8 static WMMAs).
// A tile is DMA'd global->LDS by the Tensor Data Mover (tensor_load_to_lds,
// D# per CDNA5 ISA 8.3/8.4), completion via s_wait_tensorcnt.
#define BM 64
#define BN 128
#define BK 32

template <int KTILES>
__global__ __launch_bounds__(256)
void gemm_bf16_wmma(const u16* __restrict__ A, long lda, long aBatch,
                    const u16* __restrict__ B, long ldb, long bBatch,
                    float* __restrict__ C, long ldc, long cBatch,
                    const float* __restrict__ bias,
                    int M, int N) {
    __shared__ __align__(16) u16 sA[BM * BK];       // row-major [m][k]  (TDM dest)
    __shared__ __align__(16) u16 sB[BN * BK];       // transposed [n][k]

    int b = blockIdx.z;
    A += (long)b * aBatch; B += (long)b * bBatch; C += (long)b * cBatch;

    int m0 = blockIdx.y * BM;
    int n0 = blockIdx.x * BN;
    int tid = threadIdx.x;
    int lane = tid & 31, wave = tid >> 5;
    int wm = wave & 3, wn = wave >> 2;              // 4x2 wave grid
    int lo = lane & 15, hi = lane >> 4;

    union AccU { v8f v; float f[8]; } acc[4];
    v8f z = {0.f, 0.f, 0.f, 0.f, 0.f, 0.f, 0.f, 0.f};
#pragma unroll
    for (int t = 0; t < 4; t++) acc[t].v = z;

    unsigned ldsA = (unsigned)(unsigned long long)(&sA[0]);  // LDS byte offset
    int rowsA = M - m0; if (rowsA > BM) rowsA = BM;

#pragma unroll 2
    for (int kt = 0; kt < KTILES; ++kt) {
        const int k0 = kt * BK;
        // ---- A tile 64x32 via Tensor Data Mover (wave 0 issues; EXEC ignored)
        if (wave == 0) {
            unsigned long long ga = (unsigned long long)(A + (long)m0 * lda + k0);
            v4u g0;
            g0.x = 1u;                                       // count=1, user desc
            g0.y = ldsA;                                     // lds_addr
            g0.z = (unsigned)(ga & 0xffffffffu);             // global_addr[31:0]
            g0.w = (unsigned)((ga >> 32) & 0x01ffffffu)      // global_addr[56:32]
                 | (2u << 30);                               // type=2 ("image")
            unsigned td0 = (unsigned)(KTILES * BK - k0);     // tensor_dim0 (elems)
            unsigned td1 = (unsigned)rowsA;                  // tensor_dim1 (rows)
            unsigned st0 = (unsigned)lda;                    // dim0 stride (elems)
            v8i g1;
            g1[0] = (int)(1u << 16);                         // data_size=1 -> 2B
            g1[1] = (int)((td0 & 0xffffu) << 16);            // tensor_dim0[15:0]
            g1[2] = (int)((td0 >> 16) | ((td1 & 0xffffu) << 16));
            g1[3] = (int)((td1 >> 16) | ((unsigned)BK << 16)); // tile_dim0=32
            g1[4] = (int)((unsigned)BM);                     // tile_dim1=64, dim2=0
            g1[5] = (int)st0;                                // dim0_stride[31:0]
            g1[6] = (int)(st0 >> 31 >> 1);                   // dim0_stride[47:32]
            g1[7] = 0;                                       // dim1_stride hi
            v4i zz4 = {0, 0, 0, 0};
            v8i zz8 = {0, 0, 0, 0, 0, 0, 0, 0};
            // clang-23 / therock-10.0 6-arg form:
            // (u32x4 g0, i32x8 g1, i32x4 g2, i32x4 g3, i32x8, i32 cpol)
            __builtin_amdgcn_tensor_load_to_lds(g0, g1, zz4, zz4, zz8, 0);
        }
        // ---- B tile: read 16B row chunks, store transposed [n][k]
        {
            int kk = tid >> 3;                 // 0..31
            int base = (tid & 7) * 16;         // 0..112
#pragma unroll
            for (int c = 0; c < 2; c++) {
                int cg = base + c * 8;
                uint4 val = {0u, 0u, 0u, 0u};
                if (n0 + cg < N)
                    val = *(const uint4*)(B + (long)(k0 + kk) * ldb + n0 + cg);
                const u16* e = (const u16*)&val;
#pragma unroll
                for (int j = 0; j < 8; j++) sB[(cg + j) * BK + kk] = e[j];
            }
            if (kt + 1 < KTILES && n0 + base < N)
                __builtin_prefetch(B + (long)(k0 + BK + kk) * ldb + n0 + base, 0, 1);
        }
        __builtin_amdgcn_s_wait_tensorcnt(0);
        __syncthreads();

        // A fragment: elem e -> K = hi*8 + (e<8 ? e : e+8), row = wm*16+lo
        union FragU { v16bf v; uint4 q[2]; } fa, fb;
        int arow = wm * 16 + lo;
        fa.q[0] = *(const uint4*)(sA + arow * BK + hi * 8);
        fa.q[1] = *(const uint4*)(sA + arow * BK + hi * 8 + 16);
#pragma unroll
        for (int t = 0; t < 4; t++) {
            // B fragment: elem e -> K = hi*8 + (e<8 ? e : e+8), col = lo (from transposed LDS)
            int bc = (wn * 4 + t) * 16 + lo;
            fb.q[0] = *(const uint4*)(sB + bc * BK + hi * 8);
            fb.q[1] = *(const uint4*)(sB + bc * BK + hi * 8 + 16);
            acc[t].v = __builtin_amdgcn_wmma_f32_16x16x32_bf16(
                false, fa.v, false, fb.v, (short)0, acc[t].v, false, false);
        }
        __syncthreads();
    }

    // C/D layout: VGPR r, lanes 0-15 -> M=r, lanes 16-31 -> M=8+r; N=lo
#pragma unroll
    for (int t = 0; t < 4; t++) {
        int col = n0 + (wn * 4 + t) * 16 + lo;
        if (col >= N) continue;
        float bv = bias ? bias[col] : 0.f;
#pragma unroll
        for (int r = 0; r < 8; r++) {
            int row = m0 + wm * 16 + hi * 8 + r;
            if (row < M) C[(long)row * ldc + col] = acc[t].f[r] + bv;
        }
    }
}

// ---------------- host orchestration ----------------
extern "C" void kernel_launch(void* const* d_in, const int* in_sizes, int n_in,
                              void* d_out, int out_size, void* d_ws, size_t ws_size,
                              hipStream_t stream) {
    (void)in_sizes; (void)n_in; (void)out_size; (void)ws_size;
    const float* q_emb   = (const float*)d_in[0];
    const float* t_emb   = (const float*)d_in[1];
    const int*   give_src= (const int*)d_in[2];
    const int*   give_dst= (const int*)d_in[3];
    const int*   i1_src  = (const int*)d_in[4];
    const int*   i1_dst  = (const int*)d_in[5];
    const int*   i2_src  = (const int*)d_in[6];
    const int*   i2_dst  = (const int*)d_in[7];
    const float* Wq_enc  = (const float*)d_in[8];
    const float* bq_enc  = (const float*)d_in[9];
    const float* Wc      = (const float*)d_in[10];
    const float* bc      = (const float*)d_in[11];
    const float* centers = (const float*)d_in[12];
    const float* kW      = (const float*)d_in[13];
    const float* kb      = (const float*)d_in[14];
    const float* qW      = (const float*)d_in[15];
    const float* qb      = (const float*)d_in[16];
    const float* vW      = (const float*)d_in[17];
    const float* vb      = (const float*)d_in[18];
    const float* rel_att = (const float*)d_in[19];
    const float* rel_msg = (const float*)d_in[20];
    const float* rel_pri = (const float*)d_in[21];
    const float* Wout    = (const float*)d_in[22];
    const float* bout    = (const float*)d_in[23];
    float* out = (float*)d_out;

    // ---- workspace carve-out ----
    char* wsp = (char*)d_ws;
    auto alloc = [&](size_t bytes) -> void* {
        void* p = (void*)wsp;
        wsp += (bytes + 255) & ~(size_t)255;
        return p;
    };
    u16* wq_bf   = (u16*)alloc((size_t)EMB * EMB * HNEED * 2);
    u16* kW_bf   = (u16*)alloc((size_t)2 * 2 * EMB * EMB * 2);
    u16* qW_bf   = (u16*)alloc((size_t)2 * 2 * EMB * EMB * 2);
    u16* vW_bf   = (u16*)alloc((size_t)2 * 2 * EMB * EMB * 2);
    u16* ratt_bf = (u16*)alloc((size_t)2 * 3 * HEADS * DK * DK * 2);
    u16* rmsg_bf = (u16*)alloc((size_t)2 * 3 * HEADS * DK * DK * 2);
    u16* wout_bf = (u16*)alloc((size_t)EMB * EMB * 2);
    u16* qemb_bf = (u16*)alloc((size_t)QLEN * EMB * 2);
    float* q_hs  = (float*)alloc((size_t)QLEN * EMB * HNEED * 4);
    float* h_q   = (float*)alloc((size_t)NQ * EMB * 4);
    float* assign= (float*)alloc((size_t)NQ * NCLUST * 4);
    float* h_tag = (float*)alloc((size_t)TLEN * EMB * 4);
    u16* hq_bf   = (u16*)alloc((size_t)NQ * EMB * 2);
    u16* ht_bf   = (u16*)alloc((size_t)TLEN * EMB * 2);
    float* k_tag = (float*)alloc((size_t)TLEN * EMB * 4);
    float* v_tag = (float*)alloc((size_t)TLEN * EMB * 4);
    float* q_tag = (float*)alloc((size_t)TLEN * EMB * 4);
    u16* kt_bf   = (u16*)alloc((size_t)TLEN * EMB * 2);
    u16* vt_bf   = (u16*)alloc((size_t)TLEN * EMB * 2);
    float* k2    = (float*)alloc((size_t)TLEN * EMB * 4);
    float* v2    = (float*)alloc((size_t)TLEN * EMB * 4);
    float* q_que = (float*)alloc((size_t)NQ * EMB * 4);
    float* t_q   = (float*)alloc((size_t)NQ * EMB * 4);
    float* t1    = (float*)alloc((size_t)TLEN * EMB * 4);
    float* t2    = (float*)alloc((size_t)TLEN * EMB * 4);
    int* gs      = (int*)alloc((size_t)EG3 * 4);
    int* gd      = (int*)alloc((size_t)EG3 * 4);
    float* score = (float*)alloc((size_t)EG3 * HEADS * 4);
    float* att   = (float*)alloc((size_t)EG3 * HEADS * 4);
    u32* mkey    = (u32*)alloc((size_t)NQ * HEADS * 4);
    float* ssum  = (float*)alloc((size_t)NQ * HEADS * 4);
    float* qheads= (float*)alloc((size_t)NQ * EMB * 4);
    float* tagsb = (float*)alloc((size_t)TLEN * EMB * 4);

    auto blocks1d = [](long n) { return (unsigned)((n + 255) / 256); };
    auto tobf = [&](const float* src, u16* dst, long n) {
        k_f32_to_bf16<<<blocks1d(n), 256, 0, stream>>>(src, dst, n);
    };
    auto gemm = [&](const u16* A, long lda, long aB, const u16* B, long ldb, long bB,
                    float* C, long ldc, long cB, const float* bias,
                    int M, int N, int K, int batch) {
        dim3 g((N + BN - 1) / BN, (M + BM - 1) / BM, batch);
        if (K == 768)
            gemm_bf16_wmma<768 / BK><<<g, 256, 0, stream>>>(A, lda, aB, B, ldb, bB,
                                                            C, ldc, cB, bias, M, N);
        else
            gemm_bf16_wmma<192 / BK><<<g, 256, 0, stream>>>(A, lda, aB, B, ldb, bB,
                                                            C, ldc, cB, bias, M, N);
    };
    auto run_rel = [&](const int* esrc, const int* edst, long E, const float* qmat,
                       float* outm, int n_dst, const float* rpri_grp) {
        k_fill_u32<<<blocks1d((long)n_dst * HEADS), 256, 0, stream>>>(mkey, 0u, (long)n_dst * HEADS);
        k_fill_u32<<<blocks1d((long)n_dst * HEADS), 256, 0, stream>>>((u32*)ssum, 0u, (long)n_dst * HEADS);
        k_fill_u32<<<blocks1d((long)n_dst * EMB), 256, 0, stream>>>((u32*)outm, 0u, (long)n_dst * EMB);
        k_edge_score<<<blocks1d(E * HEADS), 256, 0, stream>>>(esrc, edst, E, qmat, k2,
                                                              rpri_grp, score, mkey);
        k_edge_exp<<<blocks1d(E * HEADS), 256, 0, stream>>>(edst, E, score, mkey, att, ssum);
        k_edge_scatter<<<blocks1d(E * EMB), 256, 0, stream>>>(esrc, edst, E, att, ssum, v2, outm);
    };

    // ---- one-time conversions & edge replication ----
    tobf(Wq_enc,  wq_bf,   (long)EMB * EMB * HNEED);
    tobf(kW,      kW_bf,   (long)2 * 2 * EMB * EMB);
    tobf(qW,      qW_bf,   (long)2 * 2 * EMB * EMB);
    tobf(vW,      vW_bf,   (long)2 * 2 * EMB * EMB);
    tobf(rel_att, ratt_bf, (long)2 * 3 * HEADS * DK * DK);
    tobf(rel_msg, rmsg_bf, (long)2 * 3 * HEADS * DK * DK);
    tobf(Wout,    wout_bf, (long)EMB * EMB);
    tobf(q_emb,   qemb_bf, (long)QLEN * EMB);
    k_make_edges<<<blocks1d(EG3), 256, 0, stream>>>(give_src, give_dst, gs, gd);

    // ---- question encoder ----
    gemm(qemb_bf, EMB, 0, wq_bf, EMB * HNEED, 0, q_hs, EMB * HNEED, 0, bq_enc,
         QLEN, EMB * HNEED, EMB, 1);
    k_build_qh<<<blocks1d((long)NQ * EMB), 256, 0, stream>>>(q_hs, q_emb, h_q);
    k_assign<<<blocks1d(NQ), 256, 0, stream>>>(h_q, Wc, bc, assign);
    k_centers_add<<<blocks1d((long)NQ * EMB), 256, 0, stream>>>(assign, centers, h_q);
    k_copy_f32<<<blocks1d((long)TLEN * EMB), 256, 0, stream>>>(t_emb, h_tag, (long)TLEN * EMB);

    // ---- HGT layers ----
    for (int l = 0; l < 2; l++) {
        tobf(h_tag, ht_bf, (long)TLEN * EMB);
        gemm(ht_bf, EMB, 0, kW_bf + (size_t)(l * 2 + 0) * EMB * EMB, EMB, 0,
             k_tag, EMB, 0, kb + (l * 2 + 0) * EMB, TLEN, EMB, EMB, 1);
        gemm(ht_bf, EMB, 0, vW_bf + (size_t)(l * 2 + 0) * EMB * EMB, EMB, 0,
             v_tag, EMB, 0, vb + (l * 2 + 0) * EMB, TLEN, EMB, EMB, 1);
        gemm(ht_bf, EMB, 0, qW_bf + (size_t)(l * 2 + 0) * EMB * EMB, EMB, 0,
             q_tag, EMB, 0, qb + (l * 2 + 0) * EMB, TLEN, EMB, EMB, 1);
        tobf(h_q, hq_bf, (long)NQ * EMB);
        gemm(hq_bf, EMB, 0, qW_bf + (size_t)(l * 2 + 1) * EMB * EMB, EMB, 0,
             q_que, EMB, 0, qb + (l * 2 + 1) * EMB, NQ, EMB, EMB, 1);

        tobf(k_tag, kt_bf, (long)TLEN * EMB);
        tobf(v_tag, vt_bf, (long)TLEN * EMB);

        for (int r = 0; r < 3; r++) {
            const u16* ra = ratt_bf + (size_t)(l * 3 + r) * HEADS * DK * DK;
            const u16* rm = rmsg_bf + (size_t)(l * 3 + r) * HEADS * DK * DK;
            // per-head relation transforms: batched over gridDim.z = 4 heads
            gemm(kt_bf, EMB, DK, ra, DK, (long)DK * DK, k2, EMB, DK, nullptr,
                 TLEN, DK, DK, HEADS);
            gemm(vt_bf, EMB, DK, rm, DK, (long)DK * DK, v2, EMB, DK, nullptr,
                 TLEN, DK, DK, HEADS);
            const float* rp = rel_pri + (l * 3 + r) * HEADS;
            if (r == 0)      run_rel(gs, gd, EG3, q_que, t_q, NQ, rp);
            else if (r == 1) run_rel(i1_src, i1_dst, EI, q_tag, t1, TLEN, rp);
            else             run_rel(i2_src, i2_dst, EI, q_tag, t2, TLEN, rp);
        }
        k_avg_add<<<blocks1d((long)TLEN * EMB), 256, 0, stream>>>(h_tag, t1, t2, (long)TLEN * EMB);
        k_add_inplace<<<blocks1d((long)NQ * EMB), 256, 0, stream>>>(h_q, t_q, (long)NQ * EMB);
    }

    // ---- output projection ----
    tobf(h_q, hq_bf, (long)NQ * EMB);
    gemm(hq_bf, EMB, 0, wout_bf, EMB, 0, qheads, EMB, 0, bout, NQ, EMB, EMB, 1);
    tobf(h_tag, ht_bf, (long)TLEN * EMB);
    gemm(ht_bf, EMB, 0, wout_bf, EMB, 0, tagsb, EMB, 0, bout, TLEN, EMB, EMB, 1);

    k_write_ques<<<blocks1d((long)QLEN * HNEED * EMB), 256, 0, stream>>>(qheads, out);
    k_copy_f32<<<blocks1d((long)TLEN * EMB), 256, 0, stream>>>(
        tagsb, out + (long)QLEN * HNEED * EMB, (long)TLEN * EMB);
}
